// CausalTrajectoryTransformer_19636590478004
// MI455X (gfx1250) — matrix operand order repaired and checked by reading, not compiled, not run on hardware
//
#include <hip/hip_runtime.h>
#include <hip/hip_bf16.h>

// ---------------------------------------------------------------------------
// CausalTrajectoryTransformer forward for MI455X (gfx1250, wave32, WMMA).
// GEMMs: 128x128 block tile, 8 wave32s, wave tile 64x32 = 8 v_wmma per K-step.
// LDS staging: double-buffered GLOBAL_LOAD_ASYNC_TO_LDS_B128 (ASYNCcnt) so
// the async unit fills tile k+1 while WMMAs consume tile k.
// ---------------------------------------------------------------------------

typedef _Float16 f16;
typedef _Float16 v16h __attribute__((ext_vector_type(16)));
typedef float    v8f  __attribute__((ext_vector_type(8)));
typedef int      v4i  __attribute__((ext_vector_type(4)));

constexpr int Bc   = 4;
constexpr int Sc   = 2048;
constexpr int Dc   = 512;
constexpr int Hc   = 8;
constexpr int Lc   = 4;
constexpr int DFFc = 4 * Dc;       // 2048
constexpr int CWc  = 128;
constexpr int Mtok = Bc * Sc;      // 8192 tokens

#if defined(__HIP_DEVICE_COMPILE__) && \
    __has_builtin(__builtin_amdgcn_global_load_async_to_lds_b128)
#define HAVE_ASYNC_LDS 1
#else
#define HAVE_ASYNC_LDS 0
#endif

// ---------------------------------------------------------------------------
// f32 -> f16 conversion (weights)
// ---------------------------------------------------------------------------
__global__ __launch_bounds__(256) void cvt_f32_to_f16_kernel(
    const float* __restrict__ in, f16* __restrict__ out, int n) {
  int i = blockIdx.x * 256 + threadIdx.x;
  if (i < n) out[i] = (f16)in[i];
}

// ---------------------------------------------------------------------------
// Embedding: h = x @ W_in^T + b_in + pos ; also write f16 mirror
// ---------------------------------------------------------------------------
__global__ __launch_bounds__(256) void embed_kernel(
    const float* __restrict__ x, const float* __restrict__ Win,
    const float* __restrict__ bin, const float* __restrict__ pos,
    float* __restrict__ h, f16* __restrict__ hh) {
  int idx = blockIdx.x * 256 + threadIdx.x;          // over Mtok*D
  if (idx >= Mtok * Dc) return;
  int d = idx & (Dc - 1);
  int tok = idx >> 9;                                 // /512
  int s = tok & (Sc - 1);
  float v = x[tok * 2 + 0] * Win[d * 2 + 0]
          + x[tok * 2 + 1] * Win[d * 2 + 1]
          + bin[d] + pos[(size_t)s * Dc + d];
  h[idx]  = v;
  hh[idx] = (f16)v;
}

// ---------------------------------------------------------------------------
// WMMA GEMM: C[M,N] = A_f16[M,K] @ W_f16[N,K]^T + bias[N]  (optional GELU)
// Block = 256 threads (8 waves); block tile 128x128; BK=32, double-buffered.
// Wave grid 2(M) x 4(N); wave tile 64x32 -> acc[4][2], 8 WMMA / K-step.
// ---------------------------------------------------------------------------
__global__ __launch_bounds__(256) void gemm_wmma_kernel(
    const f16* __restrict__ A, const f16* __restrict__ W,
    const float* __restrict__ bias, float* __restrict__ C,
    f16* __restrict__ Ch, int M, int N, int K, int gelu) {
  constexpr int LDT = 40;                 // 32 halves + 8 pad (80B row stride)
  __shared__ f16 As[2][128 * LDT];        // ping-pong buffers (40KB total LDS)
  __shared__ f16 Bs[2][128 * LDT];

  const int tid  = threadIdx.x;
  const int lane = tid & 31;
  const int wave = tid >> 5;
  const int m0 = blockIdx.y * 128;
  const int n0 = blockIdx.x * 128;
  const int mBase = (wave >> 2) * 64;     // 2 wave-rows of 64
  const int nBase = (wave & 3) * 32;      // 4 wave-cols of 32
  const int r16   = lane & 15;
  const int hsel  = lane >> 4;            // lane half select

  // Staging assignment: 512 chunks of 8 halves per 128x32 tile, 2 per thread.
  const int rowS0 = tid >> 2;             // chunk tid
  const int colS0 = (tid & 3) * 8;
  const int rowS1 = (tid + 256) >> 2;     // chunk tid+256
  const int colS1 = ((tid + 256) & 3) * 8;
  const f16* aSrc0 = A + (size_t)(m0 + rowS0) * K + colS0;
  const f16* aSrc1 = A + (size_t)(m0 + rowS1) * K + colS1;
  const f16* wSrc0 = W + (size_t)(n0 + rowS0) * K + colS0;
  const f16* wSrc1 = W + (size_t)(n0 + rowS1) * K + colS1;
  const int aOff0 = rowS0 * LDT + colS0;
  const int aOff1 = rowS1 * LDT + colS1;

  // Issue the 4 async (or fallback sync) tile copies for K-chunk kk -> buf bi.
  auto stage = [&](int kk, int bi) {
    const int k0 = kk * 32;
#if HAVE_ASYNC_LDS
    typedef v4i __attribute__((address_space(1)))* gptr;
    typedef v4i __attribute__((address_space(3)))* lptr;
    __builtin_amdgcn_global_load_async_to_lds_b128(
        (gptr)(aSrc0 + k0), (lptr)&As[bi][aOff0], 0, 0);
    __builtin_amdgcn_global_load_async_to_lds_b128(
        (gptr)(aSrc1 + k0), (lptr)&As[bi][aOff1], 0, 0);
    __builtin_amdgcn_global_load_async_to_lds_b128(
        (gptr)(wSrc0 + k0), (lptr)&Bs[bi][aOff0], 0, 0);
    __builtin_amdgcn_global_load_async_to_lds_b128(
        (gptr)(wSrc1 + k0), (lptr)&Bs[bi][aOff1], 0, 0);
#else
    *(uint4*)&As[bi][aOff0] = *(const uint4*)(aSrc0 + k0);
    *(uint4*)&As[bi][aOff1] = *(const uint4*)(aSrc1 + k0);
    *(uint4*)&Bs[bi][aOff0] = *(const uint4*)(wSrc0 + k0);
    *(uint4*)&Bs[bi][aOff1] = *(const uint4*)(wSrc1 + k0);
#endif
    // Cache prefetch for the chunk after this one (global_prefetch_b8).
    if (kk * 32 + 32 < K) {
      __builtin_prefetch(aSrc0 + k0 + 32, 0, 1);
      __builtin_prefetch(wSrc0 + k0 + 32, 0, 1);
    }
  };

  v8f acc[4][2] = {};
  union Frag { v16h v; uint4 u[2]; };

  const int KT = K / 32;
  stage(0, 0);                            // prologue: tile 0 -> buf 0

  for (int kt = 0; kt < KT; ++kt) {
    const int bi = kt & 1;
#if HAVE_ASYNC_LDS
#if __has_builtin(__builtin_amdgcn_s_wait_asynccnt)
    __builtin_amdgcn_s_wait_asynccnt(0);
#else
    asm volatile("s_wait_asynccnt 0" ::: "memory");
#endif
#endif
    __syncthreads();                      // tile kt resident in buf bi

    // Kick off tile kt+1 into the other buffer; async unit runs under WMMAs.
    if (kt + 1 < KT) stage(kt + 1, bi ^ 1);

    Frag a[4], b[2];
#pragma unroll
    for (int i = 0; i < 4; ++i) {
      // A 16x32 f16 fragment (ISA layout): lanes 0-15 hold K0-7,K16-23 of
      // row M=lane; lanes 16-31 hold K8-15,K24-31.
      int m = mBase + i * 16 + r16;
      a[i].u[0] = *(const uint4*)&As[bi][m * LDT + hsel * 8];
      a[i].u[1] = *(const uint4*)&As[bi][m * LDT + 16 + hsel * 8];
    }
#pragma unroll
    for (int i = 0; i < 2; ++i) {
      // B 32x16 fragment: lane n holds column n; lanes 0-15 K0-7,K16-23,
      // lanes 16-31 K8-15,K24-31 (same K order as A).
      int n = nBase + i * 16 + r16;
      b[i].u[0] = *(const uint4*)&Bs[bi][n * LDT + hsel * 8];
      b[i].u[1] = *(const uint4*)&Bs[bi][n * LDT + 16 + hsel * 8];
    }

#pragma unroll
    for (int mi = 0; mi < 4; ++mi)
#pragma unroll
      for (int ni = 0; ni < 2; ++ni)
        acc[mi][ni] = __builtin_amdgcn_wmma_f32_16x16x32_f16(
            false, a[mi].v, false, b[ni].v, (short)0, acc[mi][ni],
            false, false);
    // NOTE: a wave's ds reads of buf bi are complete before it signals the
    // next iteration's barrier (s_wait_dscnt precedes WMMA consumption), so
    // re-staging into bi at kt+2 (issued after that barrier) cannot race.
  }

  // Epilogue: C/D layout -> VGPR r holds (M=r, N=lane) / (M=r+8, N=lane-16).
#pragma unroll
  for (int mi = 0; mi < 4; ++mi) {
#pragma unroll
    for (int ni = 0; ni < 2; ++ni) {
      int col = n0 + nBase + ni * 16 + r16;
      float bv = bias ? bias[col] : 0.0f;
#pragma unroll
      for (int r = 0; r < 8; ++r) {
        int row = m0 + mBase + mi * 16 + hsel * 8 + r;
        float v = acc[mi][ni][r] + bv;
        if (gelu) v = 0.5f * v * (1.0f + erff(v * 0.70710678118f));
        size_t off = (size_t)row * N + col;
        if (C)  C[off]  = v;
        if (Ch) Ch[off] = (f16)v;
      }
    }
  }
}

// ---------------------------------------------------------------------------
// Sliding-window causal attention (window=CW, <=129 keys). One thread per
// (b, head, q). Online softmax, fp32; tiny FLOP share vs the GEMMs.
// ---------------------------------------------------------------------------
__global__ __launch_bounds__(256) void attn_kernel(
    const float* __restrict__ qkv, f16* __restrict__ oh) {
  int gid = blockIdx.x * 256 + threadIdx.x;           // B*H*S threads
  if (gid >= Bc * Hc * Sc) return;
  int i    = gid & (Sc - 1);
  int bh   = gid >> 11;                               // /2048
  int head = bh & (Hc - 1);
  int b    = bh >> 3;

  const float* qp = qkv + ((size_t)(b * Sc + i)) * (3 * Dc) + head * 64;
  float q[64];
#pragma unroll
  for (int d = 0; d < 64; ++d) q[d] = qp[d];

  float m = -1e30f, l = 0.0f;
  float acc[64];
#pragma unroll
  for (int d = 0; d < 64; ++d) acc[d] = 0.0f;

  int j0 = (i >= CWc) ? (i - CWc) : 0;                // keys in [i-CW, i]
  for (int j = j0; j <= i; ++j) {
    const float* kp = qkv + ((size_t)(b * Sc + j)) * (3 * Dc) + Dc + head * 64;
    float s = 0.0f;
#pragma unroll
    for (int d = 0; d < 64; ++d) s += q[d] * kp[d];
    s *= 0.125f;                                      // 1/sqrt(64)
    float nm = fmaxf(m, s);
    float sc = __expf(m - nm);
    float p  = __expf(s - nm);
    l = l * sc + p;
    const float* vp = kp + Dc;
#pragma unroll
    for (int d = 0; d < 64; ++d) acc[d] = acc[d] * sc + p * vp[d];
    m = nm;
  }
  float inv = 1.0f / l;
  f16* op = oh + ((size_t)(b * Sc + i)) * Dc + head * 64;
#pragma unroll
  for (int d = 0; d < 64; ++d) op[d] = (f16)(acc[d] * inv);
}

// ---------------------------------------------------------------------------
// Fused residual add + LayerNorm over D=512. One 256-thread block per token.
// ---------------------------------------------------------------------------
__global__ __launch_bounds__(256) void add_ln_kernel(
    float* __restrict__ h, const float* __restrict__ tmp,
    const float* __restrict__ g, const float* __restrict__ bta,
    f16* __restrict__ hh) {
  __shared__ float red[256];
  int row = blockIdx.x;
  int t   = threadIdx.x;
  size_t base = (size_t)row * Dc;
  float v0 = h[base + t]       + tmp[base + t];
  float v1 = h[base + t + 256] + tmp[base + t + 256];

  red[t] = v0 + v1;
  __syncthreads();
  for (int s = 128; s > 0; s >>= 1) {
    if (t < s) red[t] += red[t + s];
    __syncthreads();
  }
  float mean = red[0] * (1.0f / 512.0f);
  __syncthreads();

  float d0 = v0 - mean, d1 = v1 - mean;
  red[t] = d0 * d0 + d1 * d1;
  __syncthreads();
  for (int s = 128; s > 0; s >>= 1) {
    if (t < s) red[t] += red[t + s];
    __syncthreads();
  }
  float rstd = rsqrtf(red[0] * (1.0f / 512.0f) + 1e-5f);

  float o0 = d0 * rstd * g[t]       + bta[t];
  float o1 = d1 * rstd * g[t + 256] + bta[t + 256];
  h[base + t]        = o0;
  h[base + t + 256]  = o1;
  hh[base + t]       = (f16)o0;
  hh[base + t + 256] = (f16)o1;
}

// ---------------------------------------------------------------------------
// Head: out = h @ W_head^T + b_head ; mu = out[:,:2], ls = clip(out[:,2:]).
// ---------------------------------------------------------------------------
__global__ __launch_bounds__(256) void head_kernel(
    const float* __restrict__ h, const float* __restrict__ Wh,
    const float* __restrict__ bh, float* __restrict__ out) {
  int t = blockIdx.x * 256 + threadIdx.x;
  if (t >= Mtok) return;
  float a0 = 0.f, a1 = 0.f, a2 = 0.f, a3 = 0.f;
  const float* hp = h + (size_t)t * Dc;
  for (int d = 0; d < Dc; ++d) {
    float hv = hp[d];
    a0 += hv * Wh[d];
    a1 += hv * Wh[Dc + d];
    a2 += hv * Wh[2 * Dc + d];
    a3 += hv * Wh[3 * Dc + d];
  }
  a0 += bh[0]; a1 += bh[1]; a2 += bh[2]; a3 += bh[3];
  out[(size_t)t * 2 + 0] = a0;
  out[(size_t)t * 2 + 1] = a1;
  float s2 = fminf(fmaxf(a2, -6.0f), 1.5f);
  float s3 = fminf(fmaxf(a3, -6.0f), 1.5f);
  out[(size_t)Mtok * 2 + (size_t)t * 2 + 0] = s2;
  out[(size_t)Mtok * 2 + (size_t)t * 2 + 1] = s3;
}

// ---------------------------------------------------------------------------
// Host-side launcher
// ---------------------------------------------------------------------------
static inline dim3 grid1(int n, int blk) { return dim3((n + blk - 1) / blk); }

extern "C" void kernel_launch(void* const* d_in, const int* in_sizes, int n_in,
                              void* d_out, int out_size, void* d_ws,
                              size_t ws_size, hipStream_t stream) {
  (void)in_sizes; (void)n_in; (void)out_size; (void)ws_size;
  const float* x     = (const float*)d_in[0];
  const float* W_in  = (const float*)d_in[1];
  const float* b_in  = (const float*)d_in[2];
  const float* pos   = (const float*)d_in[3];
  const float* Wqkv  = (const float*)d_in[4];
  const float* bqkv  = (const float*)d_in[5];
  const float* Wo    = (const float*)d_in[6];
  const float* bo    = (const float*)d_in[7];
  const float* W1    = (const float*)d_in[8];
  const float* b1    = (const float*)d_in[9];
  const float* W2    = (const float*)d_in[10];
  const float* b2    = (const float*)d_in[11];
  const float* ln1_g = (const float*)d_in[12];
  const float* ln1_b = (const float*)d_in[13];
  const float* ln2_g = (const float*)d_in[14];
  const float* ln2_b = (const float*)d_in[15];
  const float* W_head = (const float*)d_in[16];
  const float* b_head = (const float*)d_in[17];
  float* out = (float*)d_out;

  // ---- workspace carve-up ----
  char* ws = (char*)d_ws;
  size_t off = 0;
  auto take = [&](size_t bytes) { char* p = ws + off; off += (bytes + 255) & ~size_t(255); return p; };
  float* h    = (float*)take((size_t)Mtok * Dc * 4);          // fp32 master activ
  f16*   hh   = (f16*)  take((size_t)Mtok * Dc * 2);          // f16 mirror
  float* qkv  = (float*)take((size_t)Mtok * 3 * Dc * 4);      // fp32 qkv
  f16*   oh   = (f16*)  take((size_t)Mtok * Dc * 2);          // attn out (f16)
  float* tmp  = (float*)take((size_t)Mtok * Dc * 4);          // gemm out pre-LN
  f16*   ffh  = (f16*)  take((size_t)Mtok * DFFc * 2);        // GELU(ffn1) f16
  f16*   wq_h = (f16*)  take((size_t)Lc * 3 * Dc * Dc * 2);
  f16*   wo_h = (f16*)  take((size_t)Lc * Dc * Dc * 2);
  f16*   w1_h = (f16*)  take((size_t)Lc * DFFc * Dc * 2);
  f16*   w2_h = (f16*)  take((size_t)Lc * Dc * DFFc * 2);

  // ---- weights -> f16 (deterministic, every call) ----
  {
    int nqkv = Lc * 3 * Dc * Dc;
    int no   = Lc * Dc * Dc;
    int n1   = Lc * DFFc * Dc;
    cvt_f32_to_f16_kernel<<<grid1(nqkv, 256), 256, 0, stream>>>(Wqkv, wq_h, nqkv);
    cvt_f32_to_f16_kernel<<<grid1(no,   256), 256, 0, stream>>>(Wo,   wo_h, no);
    cvt_f32_to_f16_kernel<<<grid1(n1,   256), 256, 0, stream>>>(W1,   w1_h, n1);
    cvt_f32_to_f16_kernel<<<grid1(n1,   256), 256, 0, stream>>>(W2,   w2_h, n1);
  }

  // ---- embed ----
  embed_kernel<<<grid1(Mtok * Dc, 256), 256, 0, stream>>>(x, W_in, b_in, pos, h, hh);

  // ---- transformer layers ----
  for (int l = 0; l < Lc; ++l) {
    // QKV: (8192,1536) = hh(8192,512) @ Wqkv_l^T
    gemm_wmma_kernel<<<dim3((3 * Dc) / 128, Mtok / 128), 256, 0, stream>>>(
        hh, wq_h + (size_t)l * 3 * Dc * Dc, bqkv + (size_t)l * 3 * Dc,
        qkv, nullptr, Mtok, 3 * Dc, Dc, 0);

    // Sliding-window attention -> f16 output
    attn_kernel<<<grid1(Bc * Hc * Sc, 256), 256, 0, stream>>>(qkv, oh);

    // O-projection: tmp = oh @ Wo_l^T + bo_l
    gemm_wmma_kernel<<<dim3(Dc / 128, Mtok / 128), 256, 0, stream>>>(
        oh, wo_h + (size_t)l * Dc * Dc, bo + (size_t)l * Dc,
        tmp, nullptr, Mtok, Dc, Dc, 0);

    // h = LN(h + tmp)
    add_ln_kernel<<<Mtok, 256, 0, stream>>>(
        h, tmp, ln1_g + (size_t)l * Dc, ln1_b + (size_t)l * Dc, hh);

    // FFN1 + exact GELU, f16-only output (K=512, N=2048)
    gemm_wmma_kernel<<<dim3(DFFc / 128, Mtok / 128), 256, 0, stream>>>(
        hh, w1_h + (size_t)l * DFFc * Dc, b1 + (size_t)l * DFFc,
        nullptr, ffh, Mtok, DFFc, Dc, 1);

    // FFN2 (K=2048, N=512)
    gemm_wmma_kernel<<<dim3(Dc / 128, Mtok / 128), 256, 0, stream>>>(
        ffh, w2_h + (size_t)l * Dc * DFFc, b2 + (size_t)l * Dc,
        tmp, nullptr, Mtok, Dc, DFFc, 0);

    // h = LN(h + tmp)
    add_ln_kernel<<<Mtok, 256, 0, stream>>>(
        h, tmp, ln2_g + (size_t)l * Dc, ln2_b + (size_t)l * Dc, hh);
  }

  // ---- head ----
  head_kernel<<<grid1(Mtok, 256), 256, 0, stream>>>(h, W_head, b_head, out);
}
